// SoftMoE8_3719441678655
// MI455X (gfx1250) — matrix operand order repaired and checked by reading, not compiled, not run on hardware
//
#include <hip/hip_runtime.h>
#include <math.h>

typedef __bf16 bf16;
typedef __attribute__((ext_vector_type(16))) __bf16 v16bf;
typedef __attribute__((ext_vector_type(8)))  float  v8f;
typedef int v4i __attribute__((vector_size(16)));

#define B_  4
#define N_  2048
#define D_  1024
#define E_  8
#define S_  256
#define H_  512
#define ES_ 2048

#define BK_  64
#define LDA_ 72   /* padded LDS row pitch (elements): 144B -> conflict-free frag rows */
#define LDB_ 72

union FragBF { v16bf v; uint4 q[2]; };

// ---------- gfx1250 async global->LDS path (guarded; fallback = ds_store staging) ----------
#if defined(__has_builtin)
#if __has_builtin(__builtin_amdgcn_global_load_async_to_lds_b128) && \
    __has_builtin(__builtin_amdgcn_s_wait_asynccnt)
#define USE_ASYNC 1
#endif
#endif

#ifdef USE_ASYNC
#warning "gfx1250: using global_load_async_to_lds_b128 + s_wait_asynccnt"
__device__ __forceinline__ void async_cp16(const bf16* g, bf16* l) {
    // builtin signature (from hipcc diagnostic): (v4i* global_src, v4i* lds_dst, imm offset, imm cpol)
    __builtin_amdgcn_global_load_async_to_lds_b128((v4i*)g, (v4i*)l, 0, 0);
}
#else
#warning "gfx1250: async-to-LDS builtins not found; falling back to ds_store staging"
#endif

// ---------------- block reductions (256 threads) ----------------
__device__ __forceinline__ float blk_sum(float v) {
    __shared__ float sm[256];
    sm[threadIdx.x] = v; __syncthreads();
    for (int o = 128; o > 0; o >>= 1) {
        if ((int)threadIdx.x < o) sm[threadIdx.x] += sm[threadIdx.x + o];
        __syncthreads();
    }
    float r = sm[0]; __syncthreads();
    return r;
}
__device__ __forceinline__ float blk_max(float v) {
    __shared__ float sm[256];
    sm[threadIdx.x] = v; __syncthreads();
    for (int o = 128; o > 0; o >>= 1) {
        if ((int)threadIdx.x < o) sm[threadIdx.x] = fmaxf(sm[threadIdx.x], sm[threadIdx.x + o]);
        __syncthreads();
    }
    float r = sm[0]; __syncthreads();
    return r;
}

// ---------------- generic bf16 WMMA GEMM (LDS double-buffered) ----------------
// C[M,Ncol] (f32 row-major) = A[M,K](bf16 rm) * B,  B given transposed: BT[Ncol,K] bf16 rm.
// Batched over blockIdx.z. Block tile 128x64, 8 waves of 32x32 (2x2 WMMA frags).
// Requires M%128==0, Ncol%64==0, K%64==0.
__global__ __launch_bounds__(256) void gemm_bf16_tn(
    const bf16* __restrict__ A, const bf16* __restrict__ BT, float* __restrict__ C,
    int Ncol, int K,
    long long sA, long long sB, int btMod, long long sC)
{
    __shared__ bf16 As[2][128 * LDA_];   // 36864 B
    __shared__ bf16 Bs[2][64  * LDB_];   // 18432 B

    const int z = blockIdx.z;
    A  += (long long)z * sA;
    BT += (long long)(btMod ? (z % btMod) : z) * sB;
    C  += (long long)z * sC;

    const int tid  = threadIdx.x;
    const int lane = tid & 31;
    const int wave = tid >> 5;
    const int wm = wave & 3;             // 0..3
    const int wn = wave >> 2;            // 0..1
    const int m0 = blockIdx.x * 128;
    const int n0 = blockIdx.y * 64;

    const int r16  = lane & 15;
    const int half = lane >> 4;          // 0/1
    const int ka = half * 8;             // A frag: 16B chunks at ka, ka+16
    const int kb = half * 8;             // B frag: 16B chunks at kb, kb+16

    const int mt = tid >> 3;             // 0..31 (row group for staging)
    const int kc = tid & 7;              // 0..7  (16B chunk in BK)

    auto stage = [&](int s, int buf) {
        const int k0 = s * BK_;
#ifdef USE_ASYNC
#pragma unroll
        for (int i = 0; i < 4; ++i) {
            const int m = mt + i * 32;
            async_cp16(A + (size_t)(m0 + m) * K + k0 + kc * 8, &As[buf][m * LDA_ + kc * 8]);
        }
#pragma unroll
        for (int i = 0; i < 2; ++i) {
            const int n = mt + i * 32;
            async_cp16(BT + (size_t)(n0 + n) * K + k0 + kc * 8, &Bs[buf][n * LDB_ + kc * 8]);
        }
#else
#pragma unroll
        for (int i = 0; i < 4; ++i) {
            const int m = mt + i * 32;
            *(uint4*)&As[buf][m * LDA_ + kc * 8] = *(const uint4*)(A + (size_t)(m0 + m) * K + k0 + kc * 8);
        }
#pragma unroll
        for (int i = 0; i < 2; ++i) {
            const int n = mt + i * 32;
            *(uint4*)&Bs[buf][n * LDB_ + kc * 8] = *(const uint4*)(BT + (size_t)(n0 + n) * K + k0 + kc * 8);
        }
#endif
    };

    v8f acc00 = {}, acc01 = {}, acc10 = {}, acc11 = {};
    const int nStages = K / BK_;
    stage(0, 0);
    for (int s = 0; s < nStages; ++s) {
        const int cur = s & 1;
#ifdef USE_ASYNC
        __builtin_amdgcn_s_wait_asynccnt(0);
#endif
        __syncthreads();                 // stage s resident in LDS for all waves
        if (s + 1 < nStages) stage(s + 1, cur ^ 1);   // overlap next stage with compute
#pragma unroll
        for (int kk = 0; kk < BK_; kk += 32) {
            FragBF fa0, fa1, fb0, fb1;
            const bf16* pa0 = &As[cur][(wm * 32 + r16)      * LDA_ + kk + ka];
            const bf16* pa1 = &As[cur][(wm * 32 + 16 + r16) * LDA_ + kk + ka];
            const bf16* pb0 = &Bs[cur][(wn * 32 + r16)      * LDB_ + kk + kb];
            const bf16* pb1 = &Bs[cur][(wn * 32 + 16 + r16) * LDB_ + kk + kb];
            fa0.q[0] = *(const uint4*)pa0;  fa0.q[1] = *(const uint4*)(pa0 + 16);
            fa1.q[0] = *(const uint4*)pa1;  fa1.q[1] = *(const uint4*)(pa1 + 16);
            fb0.q[0] = *(const uint4*)pb0;  fb0.q[1] = *(const uint4*)(pb0 + 16);
            fb1.q[0] = *(const uint4*)pb1;  fb1.q[1] = *(const uint4*)(pb1 + 16);
            acc00 = __builtin_amdgcn_wmma_f32_16x16x32_bf16(false, fa0.v, false, fb0.v, (short)0, acc00, false, false);
            acc01 = __builtin_amdgcn_wmma_f32_16x16x32_bf16(false, fa0.v, false, fb1.v, (short)0, acc01, false, false);
            acc10 = __builtin_amdgcn_wmma_f32_16x16x32_bf16(false, fa1.v, false, fb0.v, (short)0, acc10, false, false);
            acc11 = __builtin_amdgcn_wmma_f32_16x16x32_bf16(false, fa1.v, false, fb1.v, (short)0, acc11, false, false);
        }
    }

    const int cn = n0 + wn * 32 + r16;
    const int cm = m0 + wm * 32 + half * 8;
#pragma unroll
    for (int r = 0; r < 8; ++r) {
        C[(size_t)(cm + r)      * Ncol + cn     ] = acc00[r];
        C[(size_t)(cm + r)      * Ncol + cn + 16] = acc01[r];
        C[(size_t)(cm + 16 + r) * Ncol + cn     ] = acc10[r];
        C[(size_t)(cm + 16 + r) * Ncol + cn + 16] = acc11[r];
    }
}

// ---------------- prep kernels ----------------
__global__ __launch_bounds__(256) void prep_x_k(const float* __restrict__ x,
                                                bf16* __restrict__ xn, bf16* __restrict__ xT)
{
    const int row = blockIdx.x;             // b*N + n
    const int b = row / N_, n = row % N_;
    const float* xr = x + (size_t)row * D_;
    float v[4], ss = 0.f;
#pragma unroll
    for (int i = 0; i < 4; ++i) { v[i] = xr[threadIdx.x + i * 256]; ss += v[i] * v[i]; }
    ss = blk_sum(ss);
    const float inv = 1.0f / fmaxf(sqrtf(ss), 1e-12f);
#pragma unroll
    for (int i = 0; i < 4; ++i) {
        const int d = threadIdx.x + i * 256;
        xn[(size_t)row * D_ + d]          = (bf16)(v[i] * inv);
        xT[((size_t)b * D_ + d) * N_ + n] = (bf16)v[i];
    }
}

__global__ __launch_bounds__(256) void prep_mu_k(const float* __restrict__ mu,
                                                 const float* __restrict__ scale,
                                                 bf16* __restrict__ munT)
{
    const int es = blockIdx.x;
    float v[4], ss = 0.f;
#pragma unroll
    for (int i = 0; i < 4; ++i) {
        const int d = threadIdx.x + i * 256;
        v[i] = mu[(size_t)d * ES_ + es];
        ss += v[i] * v[i];
    }
    ss = blk_sum(ss);
    const float sc = scale[0] / fmaxf(sqrtf(ss), 1e-12f);
#pragma unroll
    for (int i = 0; i < 4; ++i) {
        const int d = threadIdx.x + i * 256;
        munT[(size_t)es * D_ + d] = (bf16)(v[i] * sc);
    }
}

__global__ __launch_bounds__(256) void prep_w1_k(const float* __restrict__ W1, bf16* __restrict__ W1T)
{   // W1[E,D,H] -> W1T[E,H,D]
    const size_t idx = (size_t)blockIdx.x * 256 + threadIdx.x;
    const int e = idx / ((size_t)D_ * H_);
    const size_t r = idx % ((size_t)D_ * H_);
    const int d = r / H_, h = r % H_;
    W1T[((size_t)e * H_ + h) * D_ + d] = (bf16)W1[idx];
}

__global__ __launch_bounds__(256) void prep_w2_k(const float* __restrict__ W2, bf16* __restrict__ W2T)
{   // W2[E,H,D] -> W2T[E,D,H]
    const size_t idx = (size_t)blockIdx.x * 256 + threadIdx.x;
    const int e = idx / ((size_t)H_ * D_);
    const size_t r = idx % ((size_t)H_ * D_);
    const int h = r / D_, d = r % D_;
    W2T[((size_t)e * D_ + d) * H_ + h] = (bf16)W2[idx];
}

__global__ void zero_acc_k(float* acc) { if (threadIdx.x < 8) acc[threadIdx.x] = 0.f; }

// ---------------- softmaxes + metrics ----------------
__global__ __launch_bounds__(256) void combine_softmax_k(
    const float* __restrict__ logits, bf16* __restrict__ combine,
    float* __restrict__ rfac, float* __restrict__ acc)
{
    const int row = blockIdx.x;             // b*N + n
    const float* Lr = logits + (size_t)row * ES_;
    float v[8], mx = -3.0e38f;
#pragma unroll
    for (int i = 0; i < 8; ++i) { v[i] = Lr[threadIdx.x + i * 256]; mx = fmaxf(mx, v[i]); }
    mx = blk_max(mx);
    float s = 0.f;
#pragma unroll
    for (int i = 0; i < 8; ++i) { v[i] = __expf(v[i] - mx); s += v[i]; }
    s = blk_sum(s);
    const float inv = 1.0f / s;
    bf16* Cr = combine + (size_t)row * ES_;
    float sq = 0.f;
#pragma unroll
    for (int i = 0; i < 8; ++i) {
        const float p = v[i] * inv;
        sq += p * p;
        Cr[threadIdx.x + i * 256] = (bf16)p;
    }
    sq = blk_sum(sq);
    if (threadIdx.x == 0) {
        atomicAdd(&acc[1], sq / (sq + 1e-9f));     // cw_diag
        rfac[row] = rsqrtf(sq + 1e-9f);
    }
}

__global__ __launch_bounds__(256) void colsum_k(
    const bf16* __restrict__ combine, const float* __restrict__ rfac, float* __restrict__ acc)
{
    const int b = blockIdx.y;
    const int es = blockIdx.x * 256 + threadIdx.x;
    const bf16* Cb = combine + (size_t)b * N_ * ES_ + es;
    const float* rf = rfac + (size_t)b * N_;
    float u = 0.f;
    for (int n = 0; n < N_; ++n) u += (float)Cb[(size_t)n * ES_] * rf[n];
    const float t = blk_sum(u * u);
    if (threadIdx.x == 0) atomicAdd(&acc[0], t);   // cw_total
}

__global__ __launch_bounds__(256) void dispatch_softmax_k(
    const float* __restrict__ logits, bf16* __restrict__ dispT, float* __restrict__ acc)
{
    const int b = blockIdx.y;
    const int col = blockIdx.x * 256 + threadIdx.x;          // es
    const float* Lb = logits + (size_t)b * N_ * ES_ + col;
    float mx = -3.0e38f;
    for (int n = 0; n < N_; ++n) mx = fmaxf(mx, Lb[(size_t)n * ES_]);
    float s = 0.f;
    for (int n = 0; n < N_; ++n) s += __expf(Lb[(size_t)n * ES_] - mx);
    const float inv = 1.0f / s;
    bf16* Dr = dispT + ((size_t)b * ES_ + col) * N_;
    float sq = 0.f, sp = 0.f;
    for (int n = 0; n < N_; ++n) {
        const float p = __expf(Lb[(size_t)n * ES_] - mx) * inv;
        sq += p * p; sp += p;
        Dr[n] = (bf16)p;
    }
    const float t = (sp * sp) / (sq + 1e-9f);
    atomicAdd(&acc[2], t);
    const int e = col >> 8, sidx = col & 255;
    if (e == sidx) atomicAdd(&acc[3], t);
}

// ---------------- elementwise epilogues ----------------
__global__ __launch_bounds__(256) void sib_convert_k(const float* __restrict__ si, bf16* __restrict__ sib)
{
    const size_t idx = (size_t)blockIdx.x * 256 + threadIdx.x;
    sib[idx] = (bf16)si[idx];
}

__global__ __launch_bounds__(256) void bias_gelu_k(const float* __restrict__ h1,
                                                   const float* __restrict__ b1, bf16* __restrict__ hb)
{   // h1 [B,E,S,H]
    const size_t idx = (size_t)blockIdx.x * 256 + threadIdx.x;
    const int e = (int)((idx / ((size_t)S_ * H_)) % E_);
    const int h = (int)(idx % H_);
    const float xv = h1[idx] + b1[(size_t)e * H_ + h];
    const float g = 0.5f * xv * (1.0f + erff(xv * 0.70710678118654752f));
    hb[idx] = (bf16)g;
}

__global__ __launch_bounds__(256) void sot_k(const float* __restrict__ C2,
                                             const float* __restrict__ b2, bf16* __restrict__ sot)
{   // C2 [B,ES,D] + b2 -> sot [B,D,ES]
    const size_t idx = (size_t)blockIdx.x * 256 + threadIdx.x;
    const int b = (int)(idx / ((size_t)ES_ * D_));
    const size_t r = idx % ((size_t)ES_ * D_);
    const int es = (int)(r / D_), d = (int)(r % D_);
    const int e = es >> 8;
    const float v = C2[idx] + b2[(size_t)e * D_ + d];
    sot[((size_t)b * D_ + d) * ES_ + es] = (bf16)v;
}

__global__ void finalize_k(const float* __restrict__ acc, float* __restrict__ out)
{
    if (threadIdx.x == 0) {
        const double cw_den = (double)B_ * N_ * (N_ - 1);
        const double dw_den = (double)B_ * E_ * S_ * (ES_ - 1);
        out[(size_t)B_ * N_ * D_]     = (float)((acc[0] - acc[1]) / cw_den);
        out[(size_t)B_ * N_ * D_ + 1] = (float)((acc[2] - acc[3]) / dw_den);
    }
}

// ---------------- launch ----------------
extern "C" void kernel_launch(void* const* d_in, const int* in_sizes, int n_in,
                              void* d_out, int out_size, void* d_ws, size_t ws_size,
                              hipStream_t stream)
{
    const float* x     = (const float*)d_in[0];
    // d_in[1] attn_weight unused in 'original' eval mode
    const float* mu    = (const float*)d_in[2];
    const float* scale = (const float*)d_in[3];
    const float* W1    = (const float*)d_in[4];
    const float* b1    = (const float*)d_in[5];
    const float* W2    = (const float*)d_in[6];
    const float* b2    = (const float*)d_in[7];
    float* out = (float*)d_out;

    char* w = (char*)d_ws;
    auto alloc = [&](size_t bytes) { char* p = w; w += (bytes + 255) & ~(size_t)255; return p; };

    bf16*  xn     = (bf16*)alloc((size_t)B_ * N_ * D_ * 2);         // 16 MB
    bf16*  xT     = (bf16*)alloc((size_t)B_ * D_ * N_ * 2);         // 16 MB
    bf16*  munT   = (bf16*)alloc((size_t)ES_ * D_ * 2);             //  4 MB
    bf16*  W1T    = (bf16*)alloc((size_t)E_ * H_ * D_ * 2);         //  8 MB
    bf16*  W2T    = (bf16*)alloc((size_t)E_ * D_ * H_ * 2);         //  8 MB
    float* logits = (float*)alloc((size_t)B_ * N_ * ES_ * 4);       // 64 MB (reused region)
    bf16*  comb   = (bf16*)alloc((size_t)B_ * N_ * ES_ * 2);        // 32 MB
    bf16*  dispT  = (bf16*)alloc((size_t)B_ * ES_ * N_ * 2);        // 32 MB
    bf16*  sib    = (bf16*)alloc((size_t)B_ * ES_ * D_ * 2);        // 16 MB
    bf16*  hb     = (bf16*)alloc((size_t)B_ * E_ * S_ * H_ * 2);    //  8 MB
    bf16*  sot    = (bf16*)alloc((size_t)B_ * D_ * ES_ * 2);        // 16 MB
    float* rfac   = (float*)alloc((size_t)B_ * N_ * 4);             // 32 KB
    float* acc    = (float*)alloc(256);

    // Reuse inside `logits` (64 MB): slot_in f32 (32 MB) @+0, h1 f32 (16 MB) @+32MB, C2 f32 (32 MB) @+0.
    float* slot_in_f = logits;
    float* h1        = (float*)((char*)logits + (size_t)32 * 1024 * 1024);
    float* C2        = logits;

    zero_acc_k<<<1, 32, 0, stream>>>(acc);
    prep_x_k <<<B_ * N_, 256, 0, stream>>>(x, xn, xT);
    prep_mu_k<<<ES_,     256, 0, stream>>>(mu, scale, munT);
    prep_w1_k<<<(E_ * D_ * H_) / 256, 256, 0, stream>>>(W1, W1T);
    prep_w2_k<<<(E_ * H_ * D_) / 256, 256, 0, stream>>>(W2, W2T);

    // 1) logits[b] = xn[b] (N x D) * munT^T  -> [N, ES]
    gemm_bf16_tn<<<dim3(N_ / 128, ES_ / 64, B_), 256, 0, stream>>>(
        xn, munT, logits, ES_, D_,
        (long long)N_ * D_, 0LL, 1, (long long)N_ * ES_);

    combine_softmax_k<<<B_ * N_, 256, 0, stream>>>(logits, comb, rfac, acc);
    colsum_k<<<dim3(ES_ / 256, B_), 256, 0, stream>>>(comb, rfac, acc);
    dispatch_softmax_k<<<dim3(ES_ / 256, B_), 256, 0, stream>>>(logits, dispT, acc);

    // 2) slot_in[b] = dispT[b] (ES x N) * xT[b]^T -> [ES, D]   (overwrites logits)
    gemm_bf16_tn<<<dim3(ES_ / 128, D_ / 64, B_), 256, 0, stream>>>(
        dispT, xT, slot_in_f, D_, N_,
        (long long)ES_ * N_, (long long)D_ * N_, 0, (long long)ES_ * D_);
    sib_convert_k<<<((size_t)B_ * ES_ * D_) / 256, 256, 0, stream>>>(slot_in_f, sib);

    // 3) h1[b,e] = sib[b,e] (S x D) * W1T[e]^T -> [S, H]
    gemm_bf16_tn<<<dim3(S_ / 128, H_ / 64, B_ * E_), 256, 0, stream>>>(
        sib, W1T, h1, H_, D_,
        (long long)S_ * D_, (long long)H_ * D_, E_, (long long)S_ * H_);
    bias_gelu_k<<<((size_t)B_ * E_ * S_ * H_) / 256, 256, 0, stream>>>(h1, b1, hb);

    // 4) C2[b,e] = hb[b,e] (S x H) * W2T[e]^T -> [S, D]  (overwrites slot_in_f)
    gemm_bf16_tn<<<dim3(S_ / 128, D_ / 64, B_ * E_), 256, 0, stream>>>(
        hb, W2T, C2, D_, H_,
        (long long)S_ * H_, (long long)D_ * H_, E_, (long long)S_ * D_);
    sot_k<<<((size_t)B_ * ES_ * D_) / 256, 256, 0, stream>>>(C2, b2, sot);

    // 5) out[b] = comb[b] (N x ES) * sot[b]^T -> [N, D]
    gemm_bf16_tn<<<dim3(N_ / 128, D_ / 64, B_), 256, 0, stream>>>(
        comb, sot, out, D_, ES_,
        (long long)N_ * ES_, (long long)D_ * ES_, 0, (long long)N_ * D_);

    finalize_k<<<1, 32, 0, stream>>>(acc, out);
}
